// Net_84799834292541
// MI455X (gfx1250) — matrix operand order repaired and checked by reading, not compiled, not run on hardware
//
#include <hip/hip_runtime.h>
#include <cstdint>
#include <cstddef>

// ---------------------------------------------------------------------------
// MPNN (CGConv + GRU + Set2Set) for MI455X / gfx1250, wave32 + bf16 WMMA.
// Edge-level GEMMs are algebraically factored into node-level GEMMs:
//   z@W = x_dst@W_top + x_src@W_bot  ->  one [100k,128]@[128,1024] WMMA GEMM
// per iteration + an elementwise edge kernel with fp32 atomics.
// GEMM stages the shared B panel in LDS via gfx1250 async-copy
// (global_load_async_to_lds_b128 + s_wait_asynccnt), WMMA reads B from LDS.
// ---------------------------------------------------------------------------

constexpr int NN = 100000;   // nodes
constexpr int NE = 600000;   // edges
constexpr int NG = 1024;     // graphs
constexpr int FT = 17;       // input feature dim
constexpr int DM = 128;      // hidden dim

typedef __attribute__((ext_vector_type(16))) __bf16 v16bf;
typedef __attribute__((ext_vector_type(8)))  __bf16 v8bf;
typedef __attribute__((ext_vector_type(8)))  float  v8f;

#define DEV __device__ __forceinline__

DEV float bf2f(__bf16 b) {
    unsigned short s; __builtin_memcpy(&s, &b, 2);
    return __uint_as_float(((unsigned)s) << 16);
}
DEV __bf16 f2bf(float f) {
    unsigned u = __float_as_uint(f);
    u += 0x7fffu + ((u >> 16) & 1u);           // round-to-nearest-even
    unsigned short s = (unsigned short)(u >> 16);
    __bf16 b; __builtin_memcpy(&b, &s, 2);
    return b;
}
DEV float sigmf(float x) { return 1.f / (1.f + expf(-x)); }
DEV float spf(float x)   { return (x > 20.f) ? x : log1pf(expf(x)); }  // softplus
// order-preserving float<->uint mapping (for atomicMax on signed floats)
DEV unsigned ordf(float f) {
    unsigned u = __float_as_uint(f);
    return (u & 0x80000000u) ? ~u : (u | 0x80000000u);
}
DEV float unordf(unsigned u) {
    unsigned b = (u & 0x80000000u) ? (u & 0x7fffffffu) : ~u;
    return __uint_as_float(b);
}

// ---------------------------------------------------------------------------
// WMMA GEMM:  C[M,N] = A[M,K](bf16, row-major) x Bp(pre-packed bf16 fragments)
// Block = 256 threads = 8 waves. All waves share blockIdx.y's 64-col B panel:
// the contiguous KT*16KB/4 packed panel is async-copied to LDS once per block
// (ASYNCcnt + barrier), then each wave runs a 16-row strip x 4 wmma tiles.
// A fragment layout (16-bit A 16x32, ISA 7.12.2):
//   lanes 0-15: M=lane,    elems 0-7 -> K=kt*32+0..7,  elems 8-15 -> K=+16..23
//   lanes16-31: M=lane-16, elems 0-7 -> K=kt*32+8..15, elems 8-15 -> K=+24..31
// ---------------------------------------------------------------------------
template<int STORE_BF16, int RELU, int HAS_BIAS>
__global__ __launch_bounds__(256) void k_gemm_wmma(
    const __bf16* __restrict__ A, const __bf16* __restrict__ Bp,
    const float* __restrict__ bias, void* __restrict__ Cv,
    int M, int K, int N)
{
    const int lane = threadIdx.x & 31;
    const int wv   = threadIdx.x >> 5;
    const int rowTile = blockIdx.x * 8 + wv;
    const int colBase = blockIdx.y * 64;
    const int KT   = K >> 5;                      // K == 128 for all call sites
    const int mrow = lane & 15;
    const int kb   = (lane >> 4) << 3;

    // ---- stage 4-tile B panel (KT*4096 bytes, contiguous in Bp) into LDS ----
    __shared__ __bf16 Bs[8192];                   // 16 KB (KT=4)
    {
        const char* gpan = (const char*)(Bp + (size_t)(colBase >> 4) * KT * 512);
        unsigned ldsBase = (unsigned)(uintptr_t)(&Bs[0]);
        const int sweeps = KT;                    // 256 thr * 16B = 4096 B per sweep
        for (int i = 0; i < sweeps; ++i) {
            int byteOff = ((int)threadIdx.x + i * 256) * 16;
            unsigned laddr = ldsBase + (unsigned)byteOff;
            const char* gaddr = gpan + byteOff;
            asm volatile("global_load_async_to_lds_b128 %0, %1, off"
                         :: "v"(laddr), "v"(gaddr) : "memory");
        }
        asm volatile("s_wait_asynccnt 0x0" ::: "memory");
        __syncthreads();
    }

    if (rowTile * 16 >= M) return;                // wave-uniform (EXEC all-1s for WMMA)

    v8f a0, a1, a2, a3;
#pragma unroll
    for (int i = 0; i < 8; ++i) { a0[i] = 0.f; a1[i] = 0.f; a2[i] = 0.f; a3[i] = 0.f; }

    const __bf16* __restrict__ aRow = A + (size_t)(rowTile * 16 + mrow) * K + kb;
    const __bf16* bL = &Bs[lane * 16];            // stays addrspace(3) -> ds_load

    for (int kt = 0; kt < KT; ++kt) {
        v8bf lo = *(const v8bf*)(aRow + kt * 32);
        v8bf hi = *(const v8bf*)(aRow + kt * 32 + 16);
        v16bf af;
#pragma unroll
        for (int i = 0; i < 8; ++i) { af[i] = lo[i]; af[i + 8] = hi[i]; }
        const __bf16* pb = bL + kt * 512;
        v16bf b0 = *(const v16bf*)(pb);
        v16bf b1 = *(const v16bf*)(pb + KT * 512);
        v16bf b2 = *(const v16bf*)(pb + 2 * KT * 512);
        v16bf b3 = *(const v16bf*)(pb + 3 * KT * 512);
        a0 = __builtin_amdgcn_wmma_f32_16x16x32_bf16(false, af, false, b0, (short)0, a0, false, false);
        a1 = __builtin_amdgcn_wmma_f32_16x16x32_bf16(false, af, false, b1, (short)0, a1, false, false);
        a2 = __builtin_amdgcn_wmma_f32_16x16x32_bf16(false, af, false, b2, (short)0, a2, false, false);
        a3 = __builtin_amdgcn_wmma_f32_16x16x32_bf16(false, af, false, b3, (short)0, a3, false, false);
    }

    // C/D layout: lane -> N=lane&15; VGPR r -> M = r + (lane>=16 ? 8 : 0)
    const int mBase = rowTile * 16 + ((lane >> 4) << 3);
    const int nBase = colBase + (lane & 15);
    float*  Cf = (float*)Cv;
    __bf16* Cb = (__bf16*)Cv;
#pragma unroll
    for (int t = 0; t < 4; ++t) {
        v8f acc = (t == 0) ? a0 : ((t == 1) ? a1 : ((t == 2) ? a2 : a3));
        const int n = nBase + t * 16;
        const float bv = HAS_BIAS ? bias[n] : 0.f;
#pragma unroll
        for (int r = 0; r < 8; ++r) {
            float v = acc[r] + bv;
            if (RELU) v = fmaxf(v, 0.f);
            size_t idx = (size_t)(mBase + r) * N + n;
            if (STORE_BF16) Cb[idx] = f2bf(v); else Cf[idx] = v;
        }
    }
}

// Pack weights (fp32) into bf16 B-fragment layout. B is 32x16 per tile:
// lane l -> col n = l&15; elem e -> k = kt*32 + (l>>4)*8 + (e&7) + (e>>3)*16.
// mode 0: B[k][n] = W[(k+rowOff)*N + n]        (row-major [K,N] sub-block)
// mode 1: B[k][n] = W[n*K + k]                 (use W^T; W is [N,K] row-major)
// mode 2: B[k][n] = W[k*N+n] + W[(k+K)*N+n]    (sum of two row blocks)
__global__ void k_pack_b(const float* __restrict__ W, __bf16* __restrict__ out,
                         int K, int N, int mode, int rowOff)
{
    int id = blockIdx.x * blockDim.x + threadIdx.x;
    if (id >= K * N) return;
    int e = id & 15;
    int l = (id >> 4) & 31;
    int rest = id >> 9;
    int KT = K >> 5;
    int kt = rest % KT;
    int nt = rest / KT;
    int k = kt * 32 + ((l >> 4) << 3) + (e & 7) + ((e >> 3) << 4);
    int n = nt * 16 + (l & 15);
    float v;
    if (mode == 0)      v = W[(size_t)(k + rowOff) * N + n];
    else if (mode == 1) v = W[(size_t)n * K + k];
    else                v = W[(size_t)k * N + n] + W[(size_t)(k + K) * N + n];
    out[id] = f2bf(v);
}

// -------------------------- elementwise helpers ----------------------------
__global__ void k_f2bf(const float* __restrict__ in, __bf16* __restrict__ out, int n) {
    int i = blockIdx.x * blockDim.x + threadIdx.x; if (i < n) out[i] = f2bf(in[i]);
}
__global__ void k_fill_u32(unsigned* __restrict__ p, unsigned v, int n) {
    int i = blockIdx.x * blockDim.x + threadIdx.x; if (i < n) p[i] = v;
}
__global__ void k_copy_f32(const float* __restrict__ a, float* __restrict__ b, int n) {
    int i = blockIdx.x * blockDim.x + threadIdx.x; if (i < n) b[i] = a[i];
}
__global__ void k_relu_ip(float* __restrict__ p, int n) {
    int i = blockIdx.x * blockDim.x + threadIdx.x; if (i < n) p[i] = fmaxf(p[i], 0.f);
}

// ---------------------- stage A: conv1 (FEAT=17) ---------------------------
__global__ void k_node_prod17(const float* __restrict__ x,
                              const float* __restrict__ wf, const float* __restrict__ ws,
                              float* A, float* B, float* C, float* Dp)
{
    int id = blockIdx.x * blockDim.x + threadIdx.x;
    if (id >= NN * FT) return;
    int node = id / FT, n = id % FT;
    const float* xr = x + (size_t)node * FT;
    float a = 0.f, b = 0.f, c = 0.f, d = 0.f;
#pragma unroll
    for (int k = 0; k < FT; ++k) {
        float xv = xr[k];
        a += xv * wf[k * FT + n];        b += xv * wf[(FT + k) * FT + n];
        c += xv * ws[k * FT + n];        d += xv * ws[(FT + k) * FT + n];
    }
    A[id] = a; B[id] = b; C[id] = c; Dp[id] = d;
}
__global__ void k_edge17(const int* __restrict__ src, const int* __restrict__ dst,
                         const float* __restrict__ A, const float* __restrict__ B,
                         const float* __restrict__ C, const float* __restrict__ Dp,
                         const float* __restrict__ bf, const float* __restrict__ bs,
                         float* __restrict__ agg)
{
    int e = blockIdx.x, f = threadIdx.x;
    if (f >= FT) return;
    int s = src[e], d = dst[e];
    float gf = A[d * FT + f] + B[s * FT + f] + bf[f];
    float gs = C[d * FT + f] + Dp[s * FT + f] + bs[f];
    atomicAdd(agg + d * FT + f, sigmf(gf) * spf(gs));
}
__global__ void k_lin0(const float* __restrict__ h0, const float* __restrict__ w,
                       const float* __restrict__ b, float* __restrict__ out)
{
    int id = blockIdx.x * blockDim.x + threadIdx.x;
    if (id >= NN * DM) return;
    int node = id >> 7, d = id & 127;
    float acc = b[d];
    const float* hr = h0 + (size_t)node * FT;
#pragma unroll
    for (int k = 0; k < FT; ++k) acc += hr[k] * w[k * DM + d];
    out[id] = fmaxf(acc, 0.f);
}

// ------------------- main loop: fused conv2(add)+conv3(max) edge kernel ----
// P[node, 0:128)=A2 .. [896:1024)=D3 ; block = one edge, thread = one feature
__global__ void k_edge_main(const int* __restrict__ src, const int* __restrict__ dst,
                            const __bf16* __restrict__ P,
                            const float* __restrict__ bf2, const float* __restrict__ bs2,
                            const float* __restrict__ bf3, const float* __restrict__ bs3,
                            float* __restrict__ aggAdd, unsigned* __restrict__ aggMax)
{
    int e = blockIdx.x, f = threadIdx.x;
    int s = src[e], d = dst[e];
    const __bf16* pd = P + (size_t)d * 1024;
    const __bf16* ps = P + (size_t)s * 1024;
    float gf2 = bf2f(pd[f])       + bf2f(ps[128 + f]) + bf2[f];
    float gs2 = bf2f(pd[256 + f]) + bf2f(ps[384 + f]) + bs2[f];
    atomicAdd(aggAdd + (size_t)d * DM + f, sigmf(gf2) * spf(gs2));
    float gf3 = bf2f(pd[512 + f]) + bf2f(ps[640 + f]) + bf3[f];
    float gs3 = bf2f(pd[768 + f]) + bf2f(ps[896 + f]) + bs3[f];
    float m3 = sigmf(gf3) * spf(gs3);                 // msg >= 0 always
    atomicMax(aggMax + (size_t)d * DM + f, __float_as_uint(m3));
}
// m = h + agg_add; n = h + agg_max (max buffer holds raw f32 bits, 0-init ok)
__global__ void k_combine(const float* __restrict__ h, float* __restrict__ aggA,
                          float* __restrict__ aggM, int n)
{
    int i = blockIdx.x * blockDim.x + threadIdx.x;
    if (i < n) { aggA[i] = h[i] + aggA[i]; aggM[i] = h[i] + aggM[i]; }
}
__global__ void k_gru_gate(float* __restrict__ h, const float* __restrict__ gi,
                           const float* __restrict__ gh,
                           const float* __restrict__ bih, const float* __restrict__ bhh)
{
    int id = blockIdx.x * blockDim.x + threadIdx.x;
    if (id >= NN * DM) return;
    int dd = id & 127;
    size_t b = (size_t)(id >> 7) * 384 + dd;
    float ir  = gi[b]       + bih[dd],        hr = gh[b]       + bhh[dd];
    float iz  = gi[b + 128] + bih[128 + dd],  hz = gh[b + 128] + bhh[128 + dd];
    float in_ = gi[b + 256] + bih[256 + dd],  hn = gh[b + 256] + bhh[256 + dd];
    float r = sigmf(ir + hr), z = sigmf(iz + hz);
    float nn = tanhf(in_ + r * hn);
    h[id] = (1.f - z) * nn + z * h[id];
}

// ----------------------------- Set2Set -------------------------------------
__global__ void k_lstm_gates(const float* __restrict__ qstar, const float* __restrict__ hs,
                             const float* __restrict__ wih, const float* __restrict__ whh,
                             const float* __restrict__ bih, const float* __restrict__ bhh,
                             float* __restrict__ g)
{
    int id = blockIdx.x * blockDim.x + threadIdx.x;
    if (id >= NG * 512) return;
    int row = id >> 9, o = id & 511;
    float acc = bih[o] + bhh[o];
    const float* q  = qstar + (size_t)row * 256;
    const float* wi = wih + (size_t)o * 256;
    for (int k = 0; k < 256; ++k) acc += q[k] * wi[k];
    const float* hh = hs + (size_t)row * DM;
    const float* wh = whh + (size_t)o * DM;
    for (int k = 0; k < DM; ++k) acc += hh[k] * wh[k];
    g[id] = acc;
}
__global__ void k_lstm_update(const float* __restrict__ g, float* __restrict__ hs,
                              float* __restrict__ cs)
{
    int id = blockIdx.x * blockDim.x + threadIdx.x;
    if (id >= NG * DM) return;
    int row = id >> 7, d = id & 127;
    const float* gr = g + (size_t)row * 512;
    float i = sigmf(gr[d]), f = sigmf(gr[128 + d]);
    float gg = tanhf(gr[256 + d]), o = sigmf(gr[384 + d]);
    float c = f * cs[id] + i * gg;
    cs[id] = c;
    hs[id] = o * tanhf(c);
}
__global__ void k_attn_e(const float* __restrict__ y, const float* __restrict__ hs,
                         const int* __restrict__ batch, float* __restrict__ ev)
{
    int node = blockIdx.x * blockDim.x + threadIdx.x;
    if (node >= NN) return;
    const float* yr = y + (size_t)node * DM;
    const float* q  = hs + (size_t)batch[node] * DM;
    float acc = 0.f;
    for (int k = 0; k < DM; ++k) acc += yr[k] * q[k];
    ev[node] = acc;
}
__global__ void k_emax_atomic(const float* __restrict__ ev, const int* __restrict__ batch,
                              unsigned* __restrict__ emaxu)
{
    int node = blockIdx.x * blockDim.x + threadIdx.x;
    if (node < NN) atomicMax(emaxu + batch[node], ordf(ev[node]));
}
__global__ void k_emax_decode(const unsigned* __restrict__ emaxu, float* __restrict__ emaxf)
{
    int g = blockIdx.x * blockDim.x + threadIdx.x;
    if (g < NG) {
        float m = unordf(emaxu[g]);
        emaxf[g] = __builtin_isfinite(m) ? m : 0.f;   // jnp.where(isfinite, emax, 0)
    }
}
__global__ void k_attn_a(const float* __restrict__ ev, const int* __restrict__ batch,
                         const float* __restrict__ emaxf, float* __restrict__ av,
                         float* __restrict__ denom)
{
    int node = blockIdx.x * blockDim.x + threadIdx.x;
    if (node >= NN) return;
    int b = batch[node];
    float a = expf(ev[node] - emaxf[b]);
    av[node] = a;
    atomicAdd(denom + b, a);
}
__global__ void k_attn_r(const float* __restrict__ av, const float* __restrict__ denom,
                         const int* __restrict__ batch, const float* __restrict__ y,
                         float* __restrict__ rbuf)
{
    int id = blockIdx.x * blockDim.x + threadIdx.x;
    if (id >= NN * DM) return;
    int node = id >> 7, d = id & 127;
    int b = batch[node];
    float w = av[node] / (denom[b] + 1e-16f);
    atomicAdd(rbuf + (size_t)b * DM + d, w * y[id]);
}
__global__ void k_qstar(const float* __restrict__ hs, const float* __restrict__ rbuf,
                        float* __restrict__ qstar)
{
    int id = blockIdx.x * blockDim.x + threadIdx.x;
    if (id >= NG * DM) return;
    int row = id >> 7, d = id & 127;
    qstar[(size_t)row * 256 + d]       = hs[id];
    qstar[(size_t)row * 256 + 128 + d] = rbuf[id];
}
__global__ void k_lin2(const float* __restrict__ qstar, const float* __restrict__ w,
                       const float* __restrict__ b, float* __restrict__ p17)
{
    int id = blockIdx.x * blockDim.x + threadIdx.x;
    if (id >= NG * FT) return;
    int row = id / FT, o = id % FT;
    float acc = b[o];
    const float* q = qstar + (size_t)row * 256;
    for (int k = 0; k < 256; ++k) acc += q[k] * w[k * FT + o];
    p17[id] = fmaxf(acc, 0.f);
}
__global__ void k_lin3(const float* __restrict__ p17, const float* __restrict__ w,
                       const float* __restrict__ b, float* __restrict__ out)
{
    int row = blockIdx.x * blockDim.x + threadIdx.x;
    if (row >= NG) return;
    float acc = b[0];
#pragma unroll
    for (int k = 0; k < FT; ++k) acc += p17[row * FT + k] * w[k];
    out[row] = acc;
}

// ---------------------------------------------------------------------------
static inline int cdiv(long long a, long long b) { return (int)((a + b - 1) / b); }

extern "C" void kernel_launch(void* const* d_in, const int* in_sizes, int n_in,
                              void* d_out, int out_size, void* d_ws, size_t ws_size,
                              hipStream_t stream)
{
    // ------- inputs (setup_inputs dict order) -------
    const float* x     = (const float*)d_in[0];
    const int*   eidx  = (const int*)d_in[1];
    const int*   src   = eidx;
    const int*   dstp  = eidx + NE;
    const int*   batch = (const int*)d_in[2];
    const float* c1wf = (const float*)d_in[3],  *c1bf = (const float*)d_in[4];
    const float* c1ws = (const float*)d_in[5],  *c1bs = (const float*)d_in[6];
    const float* c2wf = (const float*)d_in[7],  *c2bf = (const float*)d_in[8];
    const float* c2ws = (const float*)d_in[9],  *c2bs = (const float*)d_in[10];
    const float* c3wf = (const float*)d_in[11], *c3bf = (const float*)d_in[12];
    const float* c3ws = (const float*)d_in[13], *c3bs = (const float*)d_in[14];
    const float* l0w  = (const float*)d_in[15], *l0b  = (const float*)d_in[16];
    const float* l1w  = (const float*)d_in[17], *l1b  = (const float*)d_in[18];
    const float* g1wih = (const float*)d_in[19], *g1whh = (const float*)d_in[20];
    const float* g1bih = (const float*)d_in[21], *g1bhh = (const float*)d_in[22];
    const float* g2wih = (const float*)d_in[23], *g2whh = (const float*)d_in[24];
    const float* g2bih = (const float*)d_in[25], *g2bhh = (const float*)d_in[26];
    const float* cnw = (const float*)d_in[27], *cnb = (const float*)d_in[28];
    const float* s2swih = (const float*)d_in[29], *s2swhh = (const float*)d_in[30];
    const float* s2sbih = (const float*)d_in[31], *s2sbhh = (const float*)d_in[32];
    const float* l2w = (const float*)d_in[33], *l2b = (const float*)d_in[34];
    const float* l3w = (const float*)d_in[35], *l3b = (const float*)d_in[36];

    // ------- workspace bump allocator -------
    char* wsb = (char*)d_ws;
    size_t off = 0;
    auto alloc = [&](size_t bytes) -> char* {
        char* p = wsb + off;
        off += (bytes + 255) & ~(size_t)255;
        return p;
    };
    __bf16* Ppack = (__bf16*)alloc((size_t)8 * 16384 * 2);    // 8 sub-blocks [128,128]
    __bf16* lin1p = (__bf16*)alloc((size_t)16384 * 2);
    __bf16* connp = (__bf16*)alloc((size_t)16384 * 2);
    __bf16* wih1p = (__bf16*)alloc((size_t)49152 * 2);
    __bf16* whh1p = (__bf16*)alloc((size_t)49152 * 2);
    __bf16* wih2p = (__bf16*)alloc((size_t)49152 * 2);
    __bf16* whh2p = (__bf16*)alloc((size_t)49152 * 2);
    float* A1  = (float*)alloc((size_t)NN * FT * 4);
    float* B1  = (float*)alloc((size_t)NN * FT * 4);
    float* C1  = (float*)alloc((size_t)NN * FT * 4);
    float* D1v = (float*)alloc((size_t)NN * FT * 4);
    float* agg1 = (float*)alloc((size_t)NN * FT * 4);
    float* h    = (float*)alloc((size_t)NN * DM * 4);
    float* jj   = (float*)alloc((size_t)NN * DM * 4);
    float* aggA = (float*)alloc((size_t)NN * DM * 4);   // lin0 out / m / y
    float* aggM = (float*)alloc((size_t)NN * DM * 4);   // n (max agg)
    float* gi   = (float*)alloc((size_t)NN * 384 * 4);
    float* gh   = (float*)alloc((size_t)NN * 384 * 4);
    __bf16* ab  = (__bf16*)alloc((size_t)NN * DM * 2);
    __bf16* bb  = (__bf16*)alloc((size_t)NN * DM * 2);
    __bf16* P   = (__bf16*)alloc((size_t)NN * 1024 * 2); // 8 partial products (bf16)
    float* qstar = (float*)alloc((size_t)NG * 256 * 4);
    float* hsb   = (float*)alloc((size_t)NG * DM * 4);
    float* csb   = (float*)alloc((size_t)NG * DM * 4);
    float* g4    = (float*)alloc((size_t)NG * 512 * 4);
    float* ev    = (float*)alloc((size_t)NN * 4);
    float* av    = (float*)alloc((size_t)NN * 4);
    unsigned* emaxu = (unsigned*)alloc((size_t)NG * 4);
    float* emaxf = (float*)alloc((size_t)NG * 4);
    float* denom = (float*)alloc((size_t)NG * 4);
    float* rbuf  = (float*)alloc((size_t)NG * DM * 4);
    float* p17   = (float*)alloc((size_t)NG * FT * 4);
    (void)ws_size; (void)n_in; (void)in_sizes; (void)out_size;

    auto gemm = [&](const __bf16* Ain, const __bf16* Bp, const float* bias, void* C,
                    int M, int K, int N, int storeBf, int relu) {
        dim3 g(cdiv(M / 16, 8), N / 64), blk(256);
        if (storeBf)    k_gemm_wmma<1, 0, 0><<<g, blk, 0, stream>>>(Ain, Bp, nullptr, C, M, K, N);
        else if (relu)  k_gemm_wmma<0, 1, 1><<<g, blk, 0, stream>>>(Ain, Bp, bias,    C, M, K, N);
        else            k_gemm_wmma<0, 0, 0><<<g, blk, 0, stream>>>(Ain, Bp, nullptr, C, M, K, N);
    };

    // ------- weight packing (once per launch; deterministic) -------
    k_pack_b<<<cdiv(128 * 128, 256), 256, 0, stream>>>(l1w, lin1p, 128, 128, 0, 0);
    const float* pblk[8] = {c2wf, c2wf, c2ws, c2ws, c3wf, c3wf, c3ws, c3ws};
    const int    roff[8] = {0, 128, 0, 128, 0, 128, 0, 128};
    for (int b2 = 0; b2 < 8; ++b2)
        k_pack_b<<<cdiv(128 * 128, 256), 256, 0, stream>>>(pblk[b2], Ppack + b2 * 16384, 128, 128, 0, roff[b2]);
    k_pack_b<<<cdiv(128 * 384, 256), 256, 0, stream>>>(g1wih, wih1p, 128, 384, 1, 0);
    k_pack_b<<<cdiv(128 * 384, 256), 256, 0, stream>>>(g1whh, whh1p, 128, 384, 1, 0);
    k_pack_b<<<cdiv(128 * 384, 256), 256, 0, stream>>>(g2wih, wih2p, 128, 384, 1, 0);
    k_pack_b<<<cdiv(128 * 384, 256), 256, 0, stream>>>(g2whh, whh2p, 128, 384, 1, 0);
    k_pack_b<<<cdiv(128 * 128, 256), 256, 0, stream>>>(cnw, connp, 128, 128, 2, 0);  // cat([h,h]) -> sum blocks

    // ------- stage A: conv1 (add) + relu + lin0 + lin1 -------
    k_node_prod17<<<cdiv(NN * FT, 256), 256, 0, stream>>>(x, c1wf, c1ws, A1, B1, C1, D1v);
    k_copy_f32<<<cdiv(NN * FT, 256), 256, 0, stream>>>(x, agg1, NN * FT);   // out = x + agg
    k_edge17<<<NE, 32, 0, stream>>>(src, dstp, A1, B1, C1, D1v, c1bf, c1bs, agg1);
    k_relu_ip<<<cdiv(NN * FT, 256), 256, 0, stream>>>(agg1, NN * FT);
    k_lin0<<<cdiv(NN * DM, 256), 256, 0, stream>>>(agg1, l0w, l0b, aggA);
    k_f2bf<<<cdiv(NN * DM, 256), 256, 0, stream>>>(aggA, ab, NN * DM);
    gemm(ab, lin1p, l1b, h, NN, 128, 128, 0, 1);                            // h = relu(.@lin1+b)
    k_copy_f32<<<cdiv(NN * DM, 256), 256, 0, stream>>>(h, jj, NN * DM);     // j = out

    // ------- 3 message-passing iterations (out2 == out always) -------
    for (int it = 0; it < 3; ++it) {
        k_f2bf<<<cdiv(NN * DM, 256), 256, 0, stream>>>(h, bb, NN * DM);
        gemm(bb, Ppack, nullptr, P, NN, 128, 1024, 1, 0);                   // all 8 partials
        k_fill_u32<<<cdiv(NN * DM, 256), 256, 0, stream>>>((unsigned*)aggA, 0u, NN * DM);
        k_fill_u32<<<cdiv(NN * DM, 256), 256, 0, stream>>>((unsigned*)aggM, 0u, NN * DM);
        k_edge_main<<<NE, 128, 0, stream>>>(src, dstp, P, c2bf, c2bs, c3bf, c3bs,
                                            aggA, (unsigned*)aggM);
        k_combine<<<cdiv(NN * DM, 256), 256, 0, stream>>>(h, aggA, aggM, NN * DM);
        // GRU1: h = gru(m=aggA, h)
        k_f2bf<<<cdiv(NN * DM, 256), 256, 0, stream>>>(aggA, ab, NN * DM);
        gemm(ab, wih1p, nullptr, gi, NN, 128, 384, 0, 0);
        gemm(bb, whh1p, nullptr, gh, NN, 128, 384, 0, 0);                   // bb still bf16(h)
        k_gru_gate<<<cdiv(NN * DM, 256), 256, 0, stream>>>(h, gi, gh, g1bih, g1bhh);
        // GRU2: j = gru(n=aggM, j)
        k_f2bf<<<cdiv(NN * DM, 256), 256, 0, stream>>>(aggM, ab, NN * DM);
        k_f2bf<<<cdiv(NN * DM, 256), 256, 0, stream>>>(jj, bb, NN * DM);
        gemm(ab, wih2p, nullptr, gi, NN, 128, 384, 0, 0);
        gemm(bb, whh2p, nullptr, gh, NN, 128, 384, 0, 0);
        k_gru_gate<<<cdiv(NN * DM, 256), 256, 0, stream>>>(jj, gi, gh, g2bih, g2bhh);
    }

    // ------- connect: y = relu(cat([h,h]) @ cw + cb) -------
    k_f2bf<<<cdiv(NN * DM, 256), 256, 0, stream>>>(h, ab, NN * DM);
    gemm(ab, connp, cnb, aggA, NN, 128, 128, 0, 1);                         // y -> aggA

    // ------- Set2Set (3 steps) -------
    k_fill_u32<<<cdiv(NG * 256, 256), 256, 0, stream>>>((unsigned*)qstar, 0u, NG * 256);
    k_fill_u32<<<cdiv(NG * DM, 256), 256, 0, stream>>>((unsigned*)hsb, 0u, NG * DM);
    k_fill_u32<<<cdiv(NG * DM, 256), 256, 0, stream>>>((unsigned*)csb, 0u, NG * DM);
    for (int s = 0; s < 3; ++s) {
        k_lstm_gates<<<cdiv(NG * 512, 256), 256, 0, stream>>>(qstar, hsb, s2swih, s2swhh,
                                                              s2sbih, s2sbhh, g4);
        k_lstm_update<<<cdiv(NG * DM, 256), 256, 0, stream>>>(g4, hsb, csb);
        k_attn_e<<<cdiv(NN, 256), 256, 0, stream>>>(aggA, hsb, batch, ev);
        k_fill_u32<<<cdiv(NG, 256), 256, 0, stream>>>(emaxu, 0x007FFFFFu, NG); // ord(-inf)
        k_emax_atomic<<<cdiv(NN, 256), 256, 0, stream>>>(ev, batch, emaxu);
        k_emax_decode<<<cdiv(NG, 256), 256, 0, stream>>>(emaxu, emaxf);
        k_fill_u32<<<cdiv(NG, 256), 256, 0, stream>>>((unsigned*)denom, 0u, NG);
        k_attn_a<<<cdiv(NN, 256), 256, 0, stream>>>(ev, batch, emaxf, av, denom);
        k_fill_u32<<<cdiv(NG * DM, 256), 256, 0, stream>>>((unsigned*)rbuf, 0u, NG * DM);
        k_attn_r<<<cdiv(NN * DM, 256), 256, 0, stream>>>(av, denom, batch, aggA, rbuf);
        k_qstar<<<cdiv(NG * DM, 256), 256, 0, stream>>>(hsb, rbuf, qstar);
    }

    // ------- head -------
    k_lin2<<<cdiv(NG * FT, 256), 256, 0, stream>>>(qstar, l2w, l2b, p17);
    k_lin3<<<cdiv(NG, 256), 256, 0, stream>>>(p17, l3w, l3b, (float*)d_out);
}
